// MixedFrequencyAttention_64699387347543
// MI455X (gfx1250) — matrix operand order repaired and checked by reading, not compiled, not run on hardware
//
#include <hip/hip_runtime.h>
#include <math.h>

// ---------------------------------------------------------------------------
// MixedFrequencyAttention for MI455X (gfx1250, wave32, WMMA bf16)
// B=2, HS=48 (POS=4608), C=512 (c2=256), HEADS=8, d=32, K=7 neighborhood.
// GEMMs via v_wmma_f32_16x16x32_bf16 with register-double-buffered fragments;
// attention via wave32 VALU + shfl_xor online softmax.
// ---------------------------------------------------------------------------

typedef __bf16  bf16x8  __attribute__((ext_vector_type(8)));
typedef __bf16  v16bf   __attribute__((ext_vector_type(16)));
typedef float   v8f     __attribute__((ext_vector_type(8)));
typedef float   f32x4   __attribute__((ext_vector_type(4)));

#define POS   4608      // B * 48 * 48
#define SPB   2304      // 48 * 48 per batch
#define HS    48
#define C2    256
#define HEADS 8
#define DHEAD 32
#define KNB   7
#define BN_EPS 1e-5f

// ---------------- input prep: fp32 -> bf16, layout fixes ----------------

__global__ void convert_sal_kernel(const float* __restrict__ sal,
                                   __bf16* __restrict__ lo,
                                   __bf16* __restrict__ hi) {
    int i = blockIdx.x * blockDim.x + threadIdx.x;   // over POS*512
    if (i >= POS * 512) return;
    int pos = i >> 9;
    int c   = i & 511;
    float v = sal[i];
    if (c < C2) lo[pos * C2 + c]        = (__bf16)v;
    else        hi[pos * C2 + (c - C2)] = (__bf16)v;
}

__global__ void convert_edge_kernel(const float* __restrict__ edge, // [B,256,48,48]
                                    __bf16* __restrict__ dst) {     // [POS,256]
    int i = blockIdx.x * blockDim.x + threadIdx.x;   // over POS*256
    if (i >= POS * C2) return;
    int pos = i >> 8;
    int c   = i & 255;
    int b   = pos / SPB;
    int s   = pos % SPB;
    dst[i] = (__bf16)edge[((size_t)(b * C2 + c)) * SPB + s];
}

// weight stage: trans=1 -> dst[n*K+k] = src[k*N+n] ; trans=0 -> straight copy
__global__ void transconv_kernel(const float* __restrict__ src,
                                 __bf16* __restrict__ dst,
                                 int K, int N, int trans, int total) {
    int i = blockIdx.x * blockDim.x + threadIdx.x;
    if (i >= total) return;
    if (trans) {
        int n = i / K, k = i - n * K;
        dst[i] = (__bf16)src[(size_t)k * N + n];
    } else {
        dst[i] = (__bf16)src[i];
    }
}

// ---------------- WMMA GEMM: C[M,N] = A[M,K](bf16) * Bt[N,K](bf16) ----------

static __device__ __forceinline__ v16bf load_frag16(const __bf16* p0,
                                                    const __bf16* p1) {
    bf16x8 a = *reinterpret_cast<const bf16x8*>(p0);
    bf16x8 b = *reinterpret_cast<const bf16x8*>(p1);
    return __builtin_shufflevector(a, b, 0, 1, 2, 3, 4, 5, 6, 7,
                                         8, 9, 10, 11, 12, 13, 14, 15);
}

// mode 0: store bf16 row-major with (ldc, coff), bias add
// mode 1: final conv epilogue: +bias, BN, ReLU, vectorized bchw fp32 store
__global__ __launch_bounds__(256)
void wmma_gemm_kernel(const __bf16* __restrict__ A,
                      const __bf16* __restrict__ Bt,
                      const float*  __restrict__ bias,
                      int M, int N, int K,
                      __bf16* __restrict__ Cbf, int ldc, int coff,
                      float* __restrict__ outF,
                      const float* __restrict__ gamma,
                      const float* __restrict__ beta,
                      const float* __restrict__ mean,
                      const float* __restrict__ var,
                      int mode) {
    int lane = threadIdx.x;            // 0..31
    int wave = threadIdx.y;            // 0..7
    int m0   = (blockIdx.x * 8 + wave) * 16;
    int nblk = blockIdx.y * 64;
    int hi   = lane >> 4;              // which half-wave
    int lm   = lane & 15;

    v8f acc[4] = {};

    const __bf16* arow = A + (size_t)(m0 + lm) * K;
    const __bf16* brow0 = Bt + (size_t)(nblk + 0 * 16 + lm) * K + hi * 16;
    const __bf16* brow1 = Bt + (size_t)(nblk + 1 * 16 + lm) * K + hi * 16;
    const __bf16* brow2 = Bt + (size_t)(nblk + 2 * 16 + lm) * K + hi * 16;
    const __bf16* brow3 = Bt + (size_t)(nblk + 3 * 16 + lm) * K + hi * 16;

    // prologue: fragments for kb = 0
    v16bf afN = load_frag16(arow + hi * 16, arow + 8 + hi * 16);
    v16bf bfN0 = load_frag16(brow0, brow0 + 8);
    v16bf bfN1 = load_frag16(brow1, brow1 + 8);
    v16bf bfN2 = load_frag16(brow2, brow2 + 8);
    v16bf bfN3 = load_frag16(brow3, brow3 + 8);

    for (int kb = 0; kb < K; kb += 32) {
        v16bf af  = afN;
        v16bf bf0 = bfN0, bf1 = bfN1, bf2 = bfN2, bf3 = bfN3;
        int kn = kb + 32;
        if (kn < K) {
            // software pipeline: issue next K-step's loads before this step's
            // WMMAs so the s_wait before the MMA group covers old loads only
            afN  = load_frag16(arow + kn + hi * 16, arow + kn + 8 + hi * 16);
            bfN0 = load_frag16(brow0 + kn, brow0 + kn + 8);
            bfN1 = load_frag16(brow1 + kn, brow1 + kn + 8);
            bfN2 = load_frag16(brow2 + kn, brow2 + kn + 8);
            bfN3 = load_frag16(brow3 + kn, brow3 + kn + 8);
        }
        acc[0] = __builtin_amdgcn_wmma_f32_16x16x32_bf16(
            false, af, false, bf0, (short)0, acc[0], false, false);
        acc[1] = __builtin_amdgcn_wmma_f32_16x16x32_bf16(
            false, af, false, bf1, (short)0, acc[1], false, false);
        acc[2] = __builtin_amdgcn_wmma_f32_16x16x32_bf16(
            false, af, false, bf2, (short)0, acc[2], false, false);
        acc[3] = __builtin_amdgcn_wmma_f32_16x16x32_bf16(
            false, af, false, bf3, (short)0, acc[3], false, false);
    }

    if (mode == 0) {
#pragma unroll
        for (int t = 0; t < 4; ++t) {
            int n = nblk + t * 16 + lm;
            float bv = bias[n];
#pragma unroll
            for (int r = 0; r < 8; ++r) {
                int m = m0 + hi * 8 + r;       // C/D layout: M = r + 8*half
                Cbf[(size_t)m * ldc + coff + n] = (__bf16)(acc[t][r] + bv);
            }
        }
    } else {
        // fused conv bias + BN + ReLU + bchw store, vectorized (8 consecutive
        // spatial positions per lane -> two 16B stores)
        int mbase = m0 + hi * 8;
        int b = mbase / SPB;
        int s0 = mbase - b * SPB;
#pragma unroll
        for (int t = 0; t < 4; ++t) {
            int n = nblk + t * 16 + lm;
            float bv = bias[n];
            float sc = rsqrtf(var[n] + BN_EPS) * gamma[n];
            float sh = beta[n] - mean[n] * sc;
            f32x4 v0, v1;
#pragma unroll
            for (int r = 0; r < 4; ++r)
                v0[r] = fmaxf((acc[t][r] + bv) * sc + sh, 0.0f);
#pragma unroll
            for (int r = 0; r < 4; ++r)
                v1[r] = fmaxf((acc[t][r + 4] + bv) * sc + sh, 0.0f);
            float* dst = outF + ((size_t)(b * 512 + n)) * SPB + s0;
            *reinterpret_cast<f32x4*>(dst)     = v0;
            *reinterpret_cast<f32x4*>(dst + 4) = v1;
        }
    }
}

// ---------------- neighborhood attention (wave32, lane = d channel) --------

__global__ __launch_bounds__(256)
void nat_attn_kernel(const __bf16* __restrict__ q,    // [POS, 256]
                     const __bf16* __restrict__ kv,   // [POS, 512] (k|v)
                     const float*  __restrict__ rpb,  // [8, 13, 13]
                     __bf16* __restrict__ aout) {     // [POS, 256]
    int lane = threadIdx.x;       // d channel, 0..31
    int head = threadIdx.y;       // 0..7
    int pos  = blockIdx.x;        // 0..4607
    int b = pos / SPB;
    int s = pos - b * SPB;
    int x = s / HS;
    int y = s - x * HS;

    const float scale = 0.17677669529663687f;   // 32^-0.5
    float qv = (float)q[(size_t)pos * C2 + head * DHEAD + lane] * scale;

    int si = x - 3; si = si < 0 ? 0 : (si > HS - KNB ? HS - KNB : si);
    int sj = y - 3; sj = sj < 0 ? 0 : (sj > HS - KNB ? HS - KNB : sj);

    float Mv  = -1e30f;
    float Sv  = 0.0f;
    float acc = 0.0f;
    const float* rp = rpb + head * (2 * KNB - 1) * (2 * KNB - 1);

    for (int u = 0; u < KNB; ++u) {
        int row  = si + u;
        int ru   = row - x + (KNB - 1);
        int rb   = b * SPB + row * HS;
        for (int v = 0; v < KNB; ++v) {
            int col = sj + v;
            size_t np = (size_t)(rb + col) * 512;
            float kval = (float)kv[np + head * DHEAD + lane];
            float p = qv * kval;
            // wave32 butterfly reduction -> all lanes hold the logit
            p += __shfl_xor(p, 16, 32);
            p += __shfl_xor(p, 8, 32);
            p += __shfl_xor(p, 4, 32);
            p += __shfl_xor(p, 2, 32);
            p += __shfl_xor(p, 1, 32);
            p += rp[ru * (2 * KNB - 1) + (col - y + (KNB - 1))];

            float vval = (float)kv[np + C2 + head * DHEAD + lane];
            float Mn = fmaxf(Mv, p);
            float rs = __expf(Mv - Mn);
            float e  = __expf(p - Mn);
            Sv  = Sv * rs + e;
            acc = acc * rs + e * vval;
            Mv  = Mn;
        }
    }
    aout[(size_t)pos * C2 + head * DHEAD + lane] = (__bf16)(acc / Sv);
}

// ---------------------------------------------------------------------------

extern "C" void kernel_launch(void* const* d_in, const int* in_sizes, int n_in,
                              void* d_out, int out_size, void* d_ws, size_t ws_size,
                              hipStream_t stream) {
    const float* sal    = (const float*)d_in[0];
    const float* edge   = (const float*)d_in[1];
    const float* sa_wq  = (const float*)d_in[2];
    const float* sa_bq  = (const float*)d_in[3];
    const float* sa_wkv = (const float*)d_in[4];
    const float* sa_bkv = (const float*)d_in[5];
    const float* sa_wo  = (const float*)d_in[6];
    const float* sa_bo  = (const float*)d_in[7];
    const float* sa_rpb = (const float*)d_in[8];
    const float* ca_wq  = (const float*)d_in[9];
    const float* ca_bq  = (const float*)d_in[10];
    const float* ca_wkv = (const float*)d_in[11];
    const float* ca_bkv = (const float*)d_in[12];
    const float* ca_wo  = (const float*)d_in[13];
    const float* ca_bo  = (const float*)d_in[14];
    const float* ca_rpb = (const float*)d_in[15];
    const float* conv_w = (const float*)d_in[16];
    const float* conv_b = (const float*)d_in[17];
    const float* gamma  = (const float*)d_in[18];
    const float* beta   = (const float*)d_in[19];
    const float* mean   = (const float*)d_in[20];
    const float* var    = (const float*)d_in[21];
    float* out = (float*)d_out;

    char* ws = (char*)d_ws;
    size_t off = 0;
    auto alloc = [&](size_t bytes) -> void* {
        void* p = ws + off;
        off += (bytes + 255) & ~(size_t)255;
        return p;
    };
    __bf16* salLo  = (__bf16*)alloc((size_t)POS * C2 * 2);
    __bf16* salHi  = (__bf16*)alloc((size_t)POS * C2 * 2);
    __bf16* edgeB  = (__bf16*)alloc((size_t)POS * C2 * 2);
    __bf16* qSa    = (__bf16*)alloc((size_t)POS * C2 * 2);
    __bf16* kvSa   = (__bf16*)alloc((size_t)POS * 512 * 2);
    __bf16* qCa    = (__bf16*)alloc((size_t)POS * C2 * 2);
    __bf16* kvCa   = (__bf16*)alloc((size_t)POS * 512 * 2);
    __bf16* aSa    = (__bf16*)alloc((size_t)POS * C2 * 2);
    __bf16* aCa    = (__bf16*)alloc((size_t)POS * C2 * 2);
    __bf16* refine = (__bf16*)alloc((size_t)POS * 512 * 2);
    __bf16* wqTsa  = (__bf16*)alloc((size_t)256 * 256 * 2);
    __bf16* wkvTsa = (__bf16*)alloc((size_t)512 * 256 * 2);
    __bf16* woTsa  = (__bf16*)alloc((size_t)256 * 256 * 2);
    __bf16* wqTca  = (__bf16*)alloc((size_t)256 * 256 * 2);
    __bf16* wkvTca = (__bf16*)alloc((size_t)512 * 256 * 2);
    __bf16* woTca  = (__bf16*)alloc((size_t)256 * 256 * 2);
    __bf16* convwB = (__bf16*)alloc((size_t)512 * 512 * 2);

    // 1) input conversion / layout
    convert_sal_kernel<<<(POS * 512) / 256, 256, 0, stream>>>(sal, salLo, salHi);
    convert_edge_kernel<<<(POS * C2) / 256, 256, 0, stream>>>(edge, edgeB);

    // 2) weight staging (bf16, N-major for B fragments)
    transconv_kernel<<<(256 * 256) / 256, 256, 0, stream>>>(sa_wq,  wqTsa,  256, 256, 1, 256 * 256);
    transconv_kernel<<<(512 * 256) / 256, 256, 0, stream>>>(sa_wkv, wkvTsa, 256, 512, 1, 512 * 256);
    transconv_kernel<<<(256 * 256) / 256, 256, 0, stream>>>(sa_wo,  woTsa,  256, 256, 1, 256 * 256);
    transconv_kernel<<<(256 * 256) / 256, 256, 0, stream>>>(ca_wq,  wqTca,  256, 256, 1, 256 * 256);
    transconv_kernel<<<(512 * 256) / 256, 256, 0, stream>>>(ca_wkv, wkvTca, 256, 512, 1, 512 * 256);
    transconv_kernel<<<(256 * 256) / 256, 256, 0, stream>>>(ca_wo,  woTca,  256, 256, 1, 256 * 256);
    transconv_kernel<<<(512 * 512) / 256, 256, 0, stream>>>(conv_w, convwB, 512, 512, 0, 512 * 512);

    dim3 blk(32, 8);
    dim3 g256(POS / 128, 256 / 64);   // N=256
    dim3 g512(POS / 128, 512 / 64);   // N=512

    // 3) projections (WMMA)
    wmma_gemm_kernel<<<g256, blk, 0, stream>>>(salLo, wqTsa,  sa_bq,  POS, 256, 256,
        qSa, 256, 0, nullptr, nullptr, nullptr, nullptr, nullptr, 0);
    wmma_gemm_kernel<<<g512, blk, 0, stream>>>(salLo, wkvTsa, sa_bkv, POS, 512, 256,
        kvSa, 512, 0, nullptr, nullptr, nullptr, nullptr, nullptr, 0);
    wmma_gemm_kernel<<<g256, blk, 0, stream>>>(edgeB, wqTca,  ca_bq,  POS, 256, 256,
        qCa, 256, 0, nullptr, nullptr, nullptr, nullptr, nullptr, 0);
    wmma_gemm_kernel<<<g512, blk, 0, stream>>>(salHi, wkvTca, ca_bkv, POS, 512, 256,
        kvCa, 512, 0, nullptr, nullptr, nullptr, nullptr, nullptr, 0);

    // 4) neighborhood attention (online softmax, wave32 per head)
    nat_attn_kernel<<<POS, blk, 0, stream>>>(qSa, kvSa, sa_rpb, aSa);
    nat_attn_kernel<<<POS, blk, 0, stream>>>(qCa, kvCa, ca_rpb, aCa);

    // 5) output projections into concat buffer refine[POS, 512]
    wmma_gemm_kernel<<<g256, blk, 0, stream>>>(aSa, woTsa, sa_bo, POS, 256, 256,
        refine, 512, 0,   nullptr, nullptr, nullptr, nullptr, nullptr, 0);
    wmma_gemm_kernel<<<g256, blk, 0, stream>>>(aCa, woTca, ca_bo, POS, 256, 256,
        refine, 512, 256, nullptr, nullptr, nullptr, nullptr, nullptr, 0);

    // 6) final 1x1 conv + BN + ReLU, fused bchw vectorized store
    wmma_gemm_kernel<<<g512, blk, 0, stream>>>(refine, convwB, conv_b, POS, 512, 512,
        nullptr, 0, 0, out, gamma, beta, mean, var, 1);

    (void)in_sizes; (void)n_in; (void)out_size; (void)ws_size;
}
